// CogMemBank_71717363908761
// MI455X (gfx1250) — matrix operand order repaired and checked by reading, not compiled, not run on hardware
//
#include <hip/hip_runtime.h>
#include <hip/hip_bf16.h>
#include <math.h>

#define BB 128
#define DD 1024
#define LL 16
#define FREQN 256
#define HALFN 128
#define DFF 4096
#define NTHREADS 256
#define NWAVES 8

typedef __attribute__((ext_vector_type(16))) __bf16 v16bf;
typedef __attribute__((ext_vector_type(8)))  __bf16 v8bf;
typedef __attribute__((ext_vector_type(8)))  float  v8f;

struct CMBParams {
  const float* tokens; const int* eids; const float* tsteps;
  const float* bank_feats; const float* bank_ts; const int* counts;
  const float* qW[2]; const float* qb[2];
  const float* kW[2]; const float* kb[2];
  const float* vW[2]; const float* vb[2];
  const float* f1W[2]; const float* f1b[2];
  const float* f2W[2]; const float* f2b[2];
  const float* ln1s[2]; const float* ln1b[2];
  const float* ln2s[2]; const float* ln2b[2];
  const float* gW; const float* gb;
  const float* m1W; const float* m1b;
  const float* m2W; const float* m2b;
  const __bf16* kT[2]; const __bf16* vT[2];
  const __bf16* m1T; const __bf16* m2T;
  float* out;
};

// ---- per-token fallback ws layout (bf16 elements) ----
#define WT_KT0  0u
#define WT_KT1  1048576u
#define WT_VT0  2097152u
#define WT_VT1  3145728u
#define WT_M2T  4194304u
#define WT_M1T  5242880u
#define WS_ELEMS 5505024ull
#define WS_NEED_BYTES (WS_ELEMS * 2ull)

// batched path total scratch need (bytes, all chunks 256B-multiples)
#define BATCH_NEED_BYTES 78406144ull

// ---- per-token kernel LDS carve (bytes) ----
#define OFF_MEMB   0
#define OFF_KINB   32768
#define OFF_SA     65536
#define OFF_SB     98304
#define OFF_EMBB   131072
#define OFF_CUR    139264
#define OFF_TMPV   143360
#define OFF_HID    147456
#define OFF_RED    163840
#define OFF_WLOG   164864
#define OFF_TVAL   164928
#define OFF_RPTR   164992
#define OFF_INTS   165120
#define SMEM_BYTES 165248

__device__ inline float block_reduce(float v, float* red) {
  int t = threadIdx.x;
  __syncthreads();
  red[t] = v;
  __syncthreads();
  for (int s = NTHREADS / 2; s > 0; s >>= 1) {
    if (t < s) red[t] += red[t + s];
    __syncthreads();
  }
  return red[0];
}

__device__ inline float silu_f(float x)    { return x / (1.f + expf(-x)); }
__device__ inline float sigmoid_f(float x) { return 1.f / (1.f + expf(-x)); }
__device__ inline float gelu_f(float x) {
  float x3 = x * x * x;
  return 0.5f * x * (1.f + tanhf(0.7978845608028654f * (x + 0.044715f * x3)));
}

// =====================================================================
// Per-token fallback GEMM (16xK @ Kx1024), A in LDS
// =====================================================================
template <int TRANSB>
__device__ void gemm16(const __bf16* __restrict__ A, int K,
                       const float* __restrict__ Bw, const __bf16* __restrict__ Bt,
                       const float* __restrict__ bias,
                       __bf16* __restrict__ dst, int mode,
                       const unsigned long long* __restrict__ rptr,
                       __bf16* __restrict__ memb, __bf16* __restrict__ kinb) {
  const int wv   = threadIdx.x >> 5;
  const int lane = threadIdx.x & 31;
  const int mrow = lane & 15;
  const int lo   = (lane < 16);
  const int kb0  = lo ? 0  : 16;
  const int kb1  = lo ? 8  : 24;
  const int kbB  = lo ? 0  : 16;
  const int ncl  = lane & 15;
  const __bf16* Ar = A + mrow * K;

  for (int nt = wv; nt < 64; nt += NWAVES) {
    const int n0 = nt * 16;
    const int nn = n0 + ncl;
    const __bf16* Btrow = TRANSB ? (Bt + (size_t)nn * K + kbB) : nullptr;
    v8f acc = {};
    for (int kk = 0; kk < K; kk += 32) {
      v16bf af, bfv;
      v8bf a0 = *(const v8bf*)(Ar + kk + kb0);
      v8bf a1 = *(const v8bf*)(Ar + kk + kb1);
#pragma unroll
      for (int i = 0; i < 8; ++i) { af[i] = a0[i]; af[8 + i] = a1[i]; }
      if (TRANSB) {
        bfv = *(const v16bf*)(Btrow + kk);
      } else {
        const float* Bp = Bw + (size_t)(kk + kbB) * DD + nn;
#pragma unroll
        for (int i = 0; i < 16; ++i) bfv[i] = (__bf16)Bp[(size_t)i * DD];
      }
      acc = __builtin_amdgcn_wmma_f32_16x16x32_bf16(false, af, false, bfv,
                                                    (short)0, acc, false, false);
    }
    const float bia = bias[nn];
#pragma unroll
    for (int i = 0; i < 8; ++i) {
      const int mm = i + (lo ? 0 : 8);
      float v = acc[i] + bia;
      if (mode == 1) v = silu_f(v);
      if (mode == 2) {
        const float* rp = (const float*)rptr[mm];
        float mv = rp[nn];
        memb[mm * DD + nn] = (__bf16)mv;
        kinb[mm * DD + nn] = (__bf16)(mv + v);
      } else {
        dst[mm * DD + nn] = (__bf16)v;
      }
    }
  }
}

// LDS-tiled transpose + f32->bf16: WT[n*K+k] = bf16(W[k*N+n])
__global__ __launch_bounds__(NTHREADS)
void transpose_bf16_kernel(const float* __restrict__ W, __bf16* __restrict__ WT,
                           int K, int N) {
  __shared__ float tile[32][33];
  const int nt = blockIdx.x * 32, kt = blockIdx.y * 32;
  const int tx = threadIdx.x & 31, ty = threadIdx.x >> 5;
  for (int r = ty; r < 32; r += 8) {
    const int k = kt + r, n = nt + tx;
    tile[r][tx] = (k < K && n < N) ? W[(size_t)k * N + n] : 0.f;
  }
  __syncthreads();
  for (int r = ty; r < 32; r += 8) {
    const int n = nt + r, k = kt + tx;
    if (n < N && k < K) WT[(size_t)n * K + k] = (__bf16)tile[tx][r];
  }
}

// =====================================================================
// Per-token monolithic kernel (fallback when ws too small for batching)
// =====================================================================
template <int TRANSB>
__global__ __launch_bounds__(NTHREADS, 1)
void cogmem_kernel(CMBParams p) {
  extern __shared__ char smem[];
  __bf16* memb = (__bf16*)(smem + OFF_MEMB);
  __bf16* kinb = (__bf16*)(smem + OFF_KINB);
  __bf16* sA   = (__bf16*)(smem + OFF_SA);
  __bf16* sB   = (__bf16*)(smem + OFF_SB);
  __bf16* embb = (__bf16*)(smem + OFF_EMBB);
  float*  cur  = (float*)(smem + OFF_CUR);
  float*  tmpv = (float*)(smem + OFF_TMPV);
  float*  hid  = (float*)(smem + OFF_HID);
  float*  red  = (float*)(smem + OFF_RED);
  float*  wlog = (float*)(smem + OFF_WLOG);
  float*  tval = (float*)(smem + OFF_TVAL);
  unsigned long long* rptr = (unsigned long long*)(smem + OFF_RPTR);
  int*    ints = (int*)(smem + OFF_INTS);

  const int b   = blockIdx.x;
  const int tid = threadIdx.x;

  if (tid == 0) {
    const int e = p.eids[b];
    int c0 = p.counts[e]; c0 = c0 < 0 ? 0 : (c0 > LL ? LL : c0);
    int* ring = ints + 1;
    int m = 0;
    for (int j = 0; j < b; ++j)
      if (p.eids[j] == e) { ring[m & (LL - 1)] = j; ++m; }
    const int total = c0 + m;
    const int cnt   = total < LL ? total : LL;
    const int start = total - cnt;
    for (int r = 0; r < LL; ++r) {
      if (r < cnt) {
        const int s = start + r;
        if (s < c0) {
          rptr[r] = (unsigned long long)(p.bank_feats + ((size_t)e * LL + s) * DD);
          tval[r] = p.bank_ts[(size_t)e * LL + s];
        } else {
          const int j = ring[(s - c0) & (LL - 1)];
          rptr[r] = (unsigned long long)(p.tokens + (size_t)j * DD);
          tval[r] = p.tsteps[j];
        }
      } else {
        rptr[r] = (unsigned long long)(p.tokens + (size_t)b * DD);
        tval[r] = 0.f;
      }
    }
    ints[0] = cnt;
  }
  for (int d = tid; d < DD; d += NTHREADS)
    cur[d] = p.tokens[(size_t)b * DD + d];
  __syncthreads();
  const int cnt = ints[0];

  if (cnt > 0) {
    const float nl2 = -13.287712379549449f / (float)HALFN;
    for (int idx = tid; idx < LL * HALFN; idx += NTHREADS) {
      const int r = idx >> 7, i = idx & (HALFN - 1);
      const float f = exp2f(nl2 * (float)i);
      const float a = tval[r] * f;
      embb[r * FREQN + i]         = (__bf16)sinf(a);
      embb[r * FREQN + HALFN + i] = (__bf16)cosf(a);
    }
    __syncthreads();
    gemm16<TRANSB>(embb, FREQN, p.m1W, p.m1T, p.m1b, sA, 1, rptr, memb, kinb);
    __syncthreads();
    gemm16<TRANSB>(sA, DD, p.m2W, p.m2T, p.m2b, nullptr, 2, rptr, memb, kinb);
    __syncthreads();

    for (int blk = 0; blk < 2; ++blk) {
      gemm16<TRANSB>(kinb, DD, p.kW[blk], p.kT[blk], p.kb[blk], sA, 0, rptr, memb, kinb);
      gemm16<TRANSB>(memb, DD, p.vW[blk], p.vT[blk], p.vb[blk], sB, 0, rptr, memb, kinb);
      __syncthreads();

      for (int n = tid; n < DD; n += NTHREADS) {
        const float* w = p.qW[blk] + n;
        float acc = p.qb[blk][n];
        for (int d = 0; d < DD; ++d) acc += cur[d] * w[(size_t)d * DD];
        tmpv[n] = acc;
      }
      __syncthreads();

      for (int k = 0; k < cnt; ++k) {
        const __bf16* kr = sA + k * DD;
        float part = 0.f;
        for (int d = tid; d < DD; d += NTHREADS) part += tmpv[d] * (float)kr[d];
        const float s = block_reduce(part, red);
        if (tid == 0) wlog[k] = s * (1.f / 32.f);
      }
      __syncthreads();
      if (tid == 0) {
        float mx = -3.4e38f;
        for (int k = 0; k < cnt; ++k) mx = fmaxf(mx, wlog[k]);
        float ssum = 0.f;
        for (int k = 0; k < cnt; ++k) { const float v = expf(wlog[k] - mx); wlog[k] = v; ssum += v; }
        const float inv = 1.f / ssum;
        for (int k = 0; k < cnt; ++k) wlog[k] *= inv;
        for (int k = cnt; k < LL; ++k) wlog[k] = 0.f;
      }
      __syncthreads();

      float psum = 0.f, psq = 0.f;
      for (int d = tid; d < DD; d += NTHREADS) {
        float a = cur[d];
        for (int k = 0; k < cnt; ++k) a += wlog[k] * (float)sB[k * DD + d];
        tmpv[d] = a; psum += a; psq += a * a;
      }
      float ssum = block_reduce(psum, red);
      float ssq  = block_reduce(psq, red);
      float mean = ssum * (1.f / DD);
      float var  = ssq * (1.f / DD) - mean * mean;
      float rstd = rsqrtf(var + 1e-6f);
      for (int d = tid; d < DD; d += NTHREADS)
        cur[d] = (tmpv[d] - mean) * rstd * p.ln1s[blk][d] + p.ln1b[blk][d];
      __syncthreads();

      for (int h = tid; h < DFF; h += NTHREADS) {
        const float* w = p.f1W[blk] + h;
        float acc = p.f1b[blk][h];
        for (int d = 0; d < DD; ++d) acc += cur[d] * w[(size_t)d * DFF];
        hid[h] = gelu_f(acc);
      }
      __syncthreads();
      psum = 0.f; psq = 0.f;
      for (int n = tid; n < DD; n += NTHREADS) {
        const float* w = p.f2W[blk] + n;
        float acc = p.f2b[blk][n];
        for (int h = 0; h < DFF; ++h) acc += hid[h] * w[(size_t)h * DD];
        acc += cur[n];
        tmpv[n] = acc; psum += acc; psq += acc * acc;
      }
      ssum = block_reduce(psum, red);
      ssq  = block_reduce(psq, red);
      mean = ssum * (1.f / DD);
      var  = ssq * (1.f / DD) - mean * mean;
      rstd = rsqrtf(var + 1e-6f);
      for (int d = tid; d < DD; d += NTHREADS)
        cur[d] = (tmpv[d] - mean) * rstd * p.ln2s[blk][d] + p.ln2b[blk][d];
      __syncthreads();
    }
  }

  const float* tok = p.tokens + (size_t)b * DD;
  for (int n = tid; n < DD; n += NTHREADS) {
    float acc = p.gb[n];
    const float* w0 = p.gW + n;
    const float* w1 = p.gW + (size_t)DD * DD + n;
    for (int d = 0; d < DD; ++d)
      acc += tok[d] * w0[(size_t)d * DD] + cur[d] * w1[(size_t)d * DD];
    const float s = sigmoid_f(acc);
    p.out[(size_t)b * DD + n] = s * tok[n] + (1.f - s) * cur[n];
  }
}

// =====================================================================
// Batched path: stage kernels over all 128 tokens at once
// =====================================================================
__global__ __launch_bounds__(BB)
void prep_kernel(const float* __restrict__ tokens, const int* __restrict__ eids,
                 const float* __restrict__ tsteps, const float* __restrict__ bank_feats,
                 const float* __restrict__ bank_ts, const int* __restrict__ counts,
                 unsigned long long* __restrict__ rptr_all, float* __restrict__ tval_all,
                 int* __restrict__ cnts) {
  __shared__ int ring[BB][LL];
  const int t = threadIdx.x;
  const int e = eids[t];
  int c0 = counts[e]; c0 = c0 < 0 ? 0 : (c0 > LL ? LL : c0);
  int m = 0;
  for (int j = 0; j < t; ++j)
    if (eids[j] == e) { ring[t][m & (LL - 1)] = j; ++m; }
  const int total = c0 + m;
  const int cnt   = total < LL ? total : LL;
  const int start = total - cnt;
  for (int r = 0; r < LL; ++r) {
    unsigned long long rp; float tv;
    if (r < cnt) {
      const int s = start + r;
      if (s < c0) {
        rp = (unsigned long long)(bank_feats + ((size_t)e * LL + s) * DD);
        tv = bank_ts[(size_t)e * LL + s];
      } else {
        const int j = ring[t][(s - c0) & (LL - 1)];
        rp = (unsigned long long)(tokens + (size_t)j * DD);
        tv = tsteps[j];
      }
    } else {
      rp = (unsigned long long)(tokens + (size_t)t * DD);
      tv = 0.f;
    }
    rptr_all[t * LL + r] = rp;
    tval_all[t * LL + r] = tv;
  }
  cnts[t] = cnt;
}

__global__ __launch_bounds__(NTHREADS)
void init_kernel(const float* __restrict__ tokens, float* __restrict__ cur,
                 __bf16* __restrict__ curb) {
  const int i = blockIdx.x * NTHREADS + threadIdx.x;    // 128*1024
  const float v = tokens[i];
  cur[i]  = v;
  curb[i] = (__bf16)v;
}

__global__ __launch_bounds__(NTHREADS)
void emb_kernel(const float* __restrict__ tval_all, __bf16* __restrict__ emb_all) {
  const int idx = blockIdx.x * NTHREADS + threadIdx.x;  // 2048*128
  const int row = idx >> 7, i = idx & (HALFN - 1);
  const float nl2 = -13.287712379549449f / (float)HALFN;
  const float f = exp2f(nl2 * (float)i);
  const float a = tval_all[row] * f;
  emb_all[(size_t)row * FREQN + i]         = (__bf16)sinf(a);
  emb_all[(size_t)row * FREQN + HALFN + i] = (__bf16)cosf(a);
}

__global__ __launch_bounds__(NTHREADS)
void attn_kernel(const int* __restrict__ cnts, const float* __restrict__ qbuf,
                 const __bf16* __restrict__ Kmat, const __bf16* __restrict__ Vmat,
                 float* __restrict__ cur, __bf16* __restrict__ curb,
                 const float* __restrict__ ln_s, const float* __restrict__ ln_b) {
  __shared__ float red[NTHREADS];
  __shared__ float wlog[LL];
  const int t = blockIdx.x, tid = threadIdx.x;
  const int cnt = cnts[t];
  const float* q = qbuf + (size_t)t * DD;
  for (int k = 0; k < cnt; ++k) {
    const __bf16* kr = Kmat + ((size_t)t * LL + k) * DD;
    float part = 0.f;
    for (int d = tid; d < DD; d += NTHREADS) part += q[d] * (float)kr[d];
    const float s = block_reduce(part, red);
    if (tid == 0) wlog[k] = s * (1.f / 32.f);
  }
  __syncthreads();
  if (tid == 0) {
    if (cnt > 0) {
      float mx = -3.4e38f;
      for (int k = 0; k < cnt; ++k) mx = fmaxf(mx, wlog[k]);
      float ss = 0.f;
      for (int k = 0; k < cnt; ++k) { const float v = expf(wlog[k] - mx); wlog[k] = v; ss += v; }
      const float inv = 1.f / ss;
      for (int k = 0; k < cnt; ++k) wlog[k] *= inv;
    }
    for (int k = (cnt < 0 ? 0 : cnt); k < LL; ++k) wlog[k] = 0.f;
  }
  __syncthreads();
  float vals[DD / NTHREADS];
  float psum = 0.f, psq = 0.f;
#pragma unroll
  for (int ii = 0; ii < DD / NTHREADS; ++ii) {
    const int d = tid + ii * NTHREADS;
    float a = cur[(size_t)t * DD + d];
    for (int k = 0; k < cnt; ++k) a += wlog[k] * (float)Vmat[((size_t)t * LL + k) * DD + d];
    vals[ii] = a; psum += a; psq += a * a;
  }
  const float ssum = block_reduce(psum, red);
  const float ssq  = block_reduce(psq, red);
  const float mean = ssum * (1.f / DD);
  const float var  = ssq * (1.f / DD) - mean * mean;
  const float rstd = rsqrtf(var + 1e-6f);
#pragma unroll
  for (int ii = 0; ii < DD / NTHREADS; ++ii) {
    const int d = tid + ii * NTHREADS;
    const float x = (vals[ii] - mean) * rstd * ln_s[d] + ln_b[d];
    cur[(size_t)t * DD + d]  = x;
    curb[(size_t)t * DD + d] = (__bf16)x;
  }
}

__global__ __launch_bounds__(NTHREADS)
void ln_kernel(const float* __restrict__ xin, float* __restrict__ cur,
               __bf16* __restrict__ curb, const float* __restrict__ ln_s,
               const float* __restrict__ ln_b) {
  __shared__ float red[NTHREADS];
  const int t = blockIdx.x, tid = threadIdx.x;
  float vals[DD / NTHREADS];
  float psum = 0.f, psq = 0.f;
#pragma unroll
  for (int ii = 0; ii < DD / NTHREADS; ++ii) {
    const int d = tid + ii * NTHREADS;
    const float a = xin[(size_t)t * DD + d];
    vals[ii] = a; psum += a; psq += a * a;
  }
  const float ssum = block_reduce(psum, red);
  const float ssq  = block_reduce(psq, red);
  const float mean = ssum * (1.f / DD);
  const float var  = ssq * (1.f / DD) - mean * mean;
  const float rstd = rsqrtf(var + 1e-6f);
#pragma unroll
  for (int ii = 0; ii < DD / NTHREADS; ++ii) {
    const int d = tid + ii * NTHREADS;
    const float x = (vals[ii] - mean) * rstd * ln_s[d] + ln_b[d];
    cur[(size_t)t * DD + d]  = x;
    curb[(size_t)t * DD + d] = (__bf16)x;
  }
}

__global__ __launch_bounds__(NTHREADS)
void sel_kernel(const int* __restrict__ cnts, const float* __restrict__ cur,
                const float* __restrict__ tokens, float* __restrict__ ret,
                __bf16* __restrict__ gA) {
  const int i = blockIdx.x * NTHREADS + threadIdx.x;    // 128*1024
  const int t = i >> 10, d = i & (DD - 1);
  const float tk = tokens[i];
  const float r  = (cnts[t] > 0) ? cur[i] : tk;
  ret[i] = r;
  gA[(size_t)t * (2 * DD) + d]      = (__bf16)tk;
  gA[(size_t)t * (2 * DD) + DD + d] = (__bf16)r;
}

// Batched GEMM: C(Mx N) = A(MxK bf16 global) @ Bt(bf16 N-major) + bias.
// grid.x = M/16 (A tile staged in LDS, +8 elem row pad), grid.y = N/1024.
// MODE: 1 silu->dstb, 2 pe(memb/kinb via rptrs), 3 bf16->dstb, 4 f32->dstf,
//       5 gelu->dstb, 6 +resid->dstf, 7 gate->outp
struct GArgs {
  const __bf16* A; const __bf16* Bt;
  const float* bias;
  __bf16* dstb; __bf16* dstb2;
  float* dstf;
  const float* resid;
  const unsigned long long* rptrs;
  const float* tok; const float* ret; float* outp;
  int K; int ldd;
};

template <int MODE>
__global__ __launch_bounds__(NTHREADS)
void gemmB_kernel(GArgs g) {
  extern __shared__ char smem[];
  __bf16* As = (__bf16*)smem;                 // 16 x (K+8)
  const int K   = g.K;
  const int lda = K + 8;
  const int m0  = blockIdx.x * 16;

  const int chunks = K / 8;                   // 16B chunks per row
  for (int idx = threadIdx.x; idx < 16 * chunks; idx += NTHREADS) {
    const int r = idx / chunks, c = idx - r * chunks;
    *(v8bf*)(As + r * lda + c * 8) = *(const v8bf*)(g.A + (size_t)(m0 + r) * K + c * 8);
  }
  __syncthreads();

  const int wv = threadIdx.x >> 5, lane = threadIdx.x & 31;
  const int mrow = lane & 15, lo = (lane < 16);
  const int kb0 = lo ? 0 : 16, kb1 = lo ? 8 : 24, kbB = lo ? 0 : 16;
  const int ncl = lane & 15;
  const __bf16* Ar = As + mrow * lda;

  for (int nt = wv; nt < 64; nt += NWAVES) {
    const int nnG = blockIdx.y * 1024 + nt * 16 + ncl;
    const __bf16* Btrow = g.Bt + (size_t)nnG * K + kbB;
    v8f acc = {};
    for (int kk = 0; kk < K; kk += 32) {
      v16bf af, bfv;
      v8bf a0 = *(const v8bf*)(Ar + kk + kb0);
      v8bf a1 = *(const v8bf*)(Ar + kk + kb1);
#pragma unroll
      for (int i = 0; i < 8; ++i) { af[i] = a0[i]; af[8 + i] = a1[i]; }
      bfv = *(const v16bf*)(Btrow + kk);
      acc = __builtin_amdgcn_wmma_f32_16x16x32_bf16(false, af, false, bfv,
                                                    (short)0, acc, false, false);
    }
    const float bia = g.bias[nnG];
#pragma unroll
    for (int i = 0; i < 8; ++i) {
      const int mm = m0 + i + (lo ? 0 : 8);
      float v = acc[i] + bia;
      if (MODE == 1) {
        g.dstb[(size_t)mm * g.ldd + nnG] = (__bf16)silu_f(v);
      } else if (MODE == 2) {
        const float* rp = (const float*)g.rptrs[mm];
        const float mv = rp[nnG];
        g.dstb [(size_t)mm * g.ldd + nnG] = (__bf16)mv;
        g.dstb2[(size_t)mm * g.ldd + nnG] = (__bf16)(mv + v);
      } else if (MODE == 3) {
        g.dstb[(size_t)mm * g.ldd + nnG] = (__bf16)v;
      } else if (MODE == 4) {
        g.dstf[(size_t)mm * g.ldd + nnG] = v;
      } else if (MODE == 5) {
        g.dstb[(size_t)mm * g.ldd + nnG] = (__bf16)gelu_f(v);
      } else if (MODE == 6) {
        g.dstf[(size_t)mm * g.ldd + nnG] = v + g.resid[(size_t)mm * g.ldd + nnG];
      } else if (MODE == 7) {
        const float s = sigmoid_f(v);
        const size_t o = (size_t)mm * DD + nnG;
        g.outp[o] = s * g.tok[o] + (1.f - s) * g.ret[o];
      }
    }
  }
}

static inline int gemmB_lds(int K) { return 16 * (K + 8) * 2; }

extern "C" void kernel_launch(void* const* d_in, const int* in_sizes, int n_in,
                              void* d_out, int out_size, void* d_ws, size_t ws_size,
                              hipStream_t stream) {
  (void)in_sizes; (void)n_in; (void)out_size;
  CMBParams P;
  P.tokens     = (const float*)d_in[0];
  P.eids       = (const int*)  d_in[1];
  P.tsteps     = (const float*)d_in[2];
  P.bank_feats = (const float*)d_in[3];
  P.bank_ts    = (const float*)d_in[4];
  P.counts     = (const int*)  d_in[5];
  for (int blk = 0; blk < 2; ++blk) {
    const int base = 6 + 14 * blk;
    P.qW[blk]   = (const float*)d_in[base + 0];
    P.qb[blk]   = (const float*)d_in[base + 1];
    P.kW[blk]   = (const float*)d_in[base + 2];
    P.kb[blk]   = (const float*)d_in[base + 3];
    P.vW[blk]   = (const float*)d_in[base + 4];
    P.vb[blk]   = (const float*)d_in[base + 5];
    P.f1W[blk]  = (const float*)d_in[base + 6];
    P.f1b[blk]  = (const float*)d_in[base + 7];
    P.f2W[blk]  = (const float*)d_in[base + 8];
    P.f2b[blk]  = (const float*)d_in[base + 9];
    P.ln1s[blk] = (const float*)d_in[base + 10];
    P.ln1b[blk] = (const float*)d_in[base + 11];
    P.ln2s[blk] = (const float*)d_in[base + 12];
    P.ln2b[blk] = (const float*)d_in[base + 13];
  }
  P.gW  = (const float*)d_in[34];
  P.gb  = (const float*)d_in[35];
  P.m1W = (const float*)d_in[36];
  P.m1b = (const float*)d_in[37];
  P.m2W = (const float*)d_in[38];
  P.m2b = (const float*)d_in[39];
  P.out = (float*)d_out;

  const dim3 tb(NTHREADS);

  if (d_ws != nullptr && ws_size >= BATCH_NEED_BYTES) {
    // ---------------- batched path ----------------
    size_t off = 0;
    char* base = (char*)d_ws;
    auto alloc = [&](size_t bytes) -> char* {
      char* p = base + off;
      off += (bytes + 255) & ~(size_t)255;
      return p;
    };
    __bf16* kT0 = (__bf16*)alloc(2097152);
    __bf16* kT1 = (__bf16*)alloc(2097152);
    __bf16* vT0 = (__bf16*)alloc(2097152);
    __bf16* vT1 = (__bf16*)alloc(2097152);
    __bf16* qT0 = (__bf16*)alloc(2097152);
    __bf16* qT1 = (__bf16*)alloc(2097152);
    __bf16* m2T = (__bf16*)alloc(2097152);
    __bf16* m1T = (__bf16*)alloc(524288);
    __bf16* f1T0 = (__bf16*)alloc(8388608);
    __bf16* f1T1 = (__bf16*)alloc(8388608);
    __bf16* f2T0 = (__bf16*)alloc(8388608);
    __bf16* f2T1 = (__bf16*)alloc(8388608);
    __bf16* gT   = (__bf16*)alloc(4194304);
    __bf16* emb_all  = (__bf16*)alloc(1048576);   // 2048 x 256
    __bf16* h_all    = (__bf16*)alloc(4194304);   // 2048 x 1024
    __bf16* memb_all = (__bf16*)alloc(4194304);
    __bf16* kinb_all = (__bf16*)alloc(4194304);
    __bf16* Kmat     = (__bf16*)alloc(4194304);
    __bf16* Vmat     = (__bf16*)alloc(4194304);
    float*  cur_all  = (float*)alloc(524288);     // 128 x 1024
    __bf16* curb     = (__bf16*)alloc(262144);
    __bf16* hid_all  = (__bf16*)alloc(1048576);   // 128 x 4096
    float*  tmpf     = (float*)alloc(524288);
    float*  ret_all  = (float*)alloc(524288);
    __bf16* gA       = (__bf16*)alloc(524288);    // 128 x 2048
    unsigned long long* rptr_all = (unsigned long long*)alloc(16384);
    float*  tval_all = (float*)alloc(8192);
    int*    cnts     = (int*)alloc(512);

    const __bf16* kT[2]  = {kT0, kT1};
    const __bf16* vT[2]  = {vT0, vT1};
    const __bf16* qT[2]  = {qT0, qT1};
    const __bf16* f1T[2] = {f1T0, f1T1};
    const __bf16* f2T[2] = {f2T0, f2T1};

    // weight staging (bf16, N-major)
    transpose_bf16_kernel<<<dim3(32, 32), tb, 0, stream>>>(P.kW[0], kT0, DD, DD);
    transpose_bf16_kernel<<<dim3(32, 32), tb, 0, stream>>>(P.kW[1], kT1, DD, DD);
    transpose_bf16_kernel<<<dim3(32, 32), tb, 0, stream>>>(P.vW[0], vT0, DD, DD);
    transpose_bf16_kernel<<<dim3(32, 32), tb, 0, stream>>>(P.vW[1], vT1, DD, DD);
    transpose_bf16_kernel<<<dim3(32, 32), tb, 0, stream>>>(P.qW[0], qT0, DD, DD);
    transpose_bf16_kernel<<<dim3(32, 32), tb, 0, stream>>>(P.qW[1], qT1, DD, DD);
    transpose_bf16_kernel<<<dim3(32, 32), tb, 0, stream>>>(P.m2W,  m2T, DD, DD);
    transpose_bf16_kernel<<<dim3(32, 8),  tb, 0, stream>>>(P.m1W,  m1T, FREQN, DD);
    transpose_bf16_kernel<<<dim3(128, 32), tb, 0, stream>>>(P.f1W[0], f1T0, DD, DFF);
    transpose_bf16_kernel<<<dim3(128, 32), tb, 0, stream>>>(P.f1W[1], f1T1, DD, DFF);
    transpose_bf16_kernel<<<dim3(32, 128), tb, 0, stream>>>(P.f2W[0], f2T0, DFF, DD);
    transpose_bf16_kernel<<<dim3(32, 128), tb, 0, stream>>>(P.f2W[1], f2T1, DFF, DD);
    transpose_bf16_kernel<<<dim3(32, 64),  tb, 0, stream>>>(P.gW, gT, 2 * DD, DD);

    prep_kernel<<<dim3(1), dim3(BB), 0, stream>>>(P.tokens, P.eids, P.tsteps,
        P.bank_feats, P.bank_ts, P.counts, rptr_all, tval_all, cnts);
    init_kernel<<<dim3(512), tb, 0, stream>>>(P.tokens, cur_all, curb);
    emb_kernel<<<dim3(1024), tb, 0, stream>>>(tval_all, emb_all);

    (void)hipFuncSetAttribute((const void*)gemmB_kernel<1>, hipFuncAttributeMaxDynamicSharedMemorySize, gemmB_lds(FREQN));
    (void)hipFuncSetAttribute((const void*)gemmB_kernel<2>, hipFuncAttributeMaxDynamicSharedMemorySize, gemmB_lds(DD));
    (void)hipFuncSetAttribute((const void*)gemmB_kernel<3>, hipFuncAttributeMaxDynamicSharedMemorySize, gemmB_lds(DD));
    (void)hipFuncSetAttribute((const void*)gemmB_kernel<4>, hipFuncAttributeMaxDynamicSharedMemorySize, gemmB_lds(DD));
    (void)hipFuncSetAttribute((const void*)gemmB_kernel<5>, hipFuncAttributeMaxDynamicSharedMemorySize, gemmB_lds(DD));
    (void)hipFuncSetAttribute((const void*)gemmB_kernel<6>, hipFuncAttributeMaxDynamicSharedMemorySize, gemmB_lds(DFF));
    (void)hipFuncSetAttribute((const void*)gemmB_kernel<7>, hipFuncAttributeMaxDynamicSharedMemorySize, gemmB_lds(2 * DD));

    GArgs ga = {};
    // embed MLP: h = silu(emb @ m1)   (M=2048)
    ga.A = emb_all; ga.Bt = m1T; ga.bias = P.m1b; ga.dstb = h_all; ga.K = FREQN; ga.ldd = DD;
    gemmB_kernel<1><<<dim3(BB * LL / 16, 1), tb, gemmB_lds(FREQN), stream>>>(ga);
    // pe = h @ m2 ; memb = mem, kinb = mem + pe
    ga = {}; ga.A = h_all; ga.Bt = m2T; ga.bias = P.m2b; ga.dstb = memb_all; ga.dstb2 = kinb_all;
    ga.rptrs = rptr_all; ga.K = DD; ga.ldd = DD;
    gemmB_kernel<2><<<dim3(BB * LL / 16, 1), tb, gemmB_lds(DD), stream>>>(ga);

    for (int blk = 0; blk < 2; ++blk) {
      ga = {}; ga.A = kinb_all; ga.Bt = kT[blk]; ga.bias = P.kb[blk]; ga.dstb = Kmat; ga.K = DD; ga.ldd = DD;
      gemmB_kernel<3><<<dim3(BB * LL / 16, 1), tb, gemmB_lds(DD), stream>>>(ga);
      ga = {}; ga.A = memb_all; ga.Bt = vT[blk]; ga.bias = P.vb[blk]; ga.dstb = Vmat; ga.K = DD; ga.ldd = DD;
      gemmB_kernel<3><<<dim3(BB * LL / 16, 1), tb, gemmB_lds(DD), stream>>>(ga);
      // q projection (M=128)
      ga = {}; ga.A = curb; ga.Bt = qT[blk]; ga.bias = P.qb[blk]; ga.dstf = tmpf; ga.K = DD; ga.ldd = DD;
      gemmB_kernel<4><<<dim3(BB / 16, 1), tb, gemmB_lds(DD), stream>>>(ga);
      attn_kernel<<<dim3(BB), tb, 0, stream>>>(cnts, tmpf, Kmat, Vmat, cur_all, curb,
                                               P.ln1s[blk], P.ln1b[blk]);
      // FFN
      ga = {}; ga.A = curb; ga.Bt = f1T[blk]; ga.bias = P.f1b[blk]; ga.dstb = hid_all; ga.K = DD; ga.ldd = DFF;
      gemmB_kernel<5><<<dim3(BB / 16, DFF / 1024), tb, gemmB_lds(DD), stream>>>(ga);
      ga = {}; ga.A = hid_all; ga.Bt = f2T[blk]; ga.bias = P.f2b[blk]; ga.dstf = tmpf; ga.resid = cur_all;
      ga.K = DFF; ga.ldd = DD;
      gemmB_kernel<6><<<dim3(BB / 16, 1), tb, gemmB_lds(DFF), stream>>>(ga);
      ln_kernel<<<dim3(BB), tb, 0, stream>>>(tmpf, cur_all, curb, P.ln2s[blk], P.ln2b[blk]);
    }

    sel_kernel<<<dim3(512), tb, 0, stream>>>(cnts, cur_all, P.tokens, ret_all, gA);
    ga = {}; ga.A = gA; ga.Bt = gT; ga.bias = P.gb; ga.tok = P.tokens; ga.ret = ret_all;
    ga.outp = P.out; ga.K = 2 * DD; ga.ldd = DD;
    gemmB_kernel<7><<<dim3(BB / 16, 1), tb, gemmB_lds(2 * DD), stream>>>(ga);
    return;
  }

  // ---------------- per-token fallback ----------------
  __bf16* ws = (__bf16*)d_ws;
  P.kT[0] = ws + WT_KT0;  P.kT[1] = ws + WT_KT1;
  P.vT[0] = ws + WT_VT0;  P.vT[1] = ws + WT_VT1;
  P.m2T   = ws + WT_M2T;  P.m1T   = ws + WT_M1T;

  const bool use_trans = (d_ws != nullptr) && (ws_size >= WS_NEED_BYTES);
  if (use_trans) {
    transpose_bf16_kernel<<<dim3(32, 32), tb, 0, stream>>>(P.kW[0], (__bf16*)P.kT[0], DD, DD);
    transpose_bf16_kernel<<<dim3(32, 32), tb, 0, stream>>>(P.kW[1], (__bf16*)P.kT[1], DD, DD);
    transpose_bf16_kernel<<<dim3(32, 32), tb, 0, stream>>>(P.vW[0], (__bf16*)P.vT[0], DD, DD);
    transpose_bf16_kernel<<<dim3(32, 32), tb, 0, stream>>>(P.vW[1], (__bf16*)P.vT[1], DD, DD);
    transpose_bf16_kernel<<<dim3(32, 32), tb, 0, stream>>>(P.m2W,  (__bf16*)P.m2T,  DD, DD);
    transpose_bf16_kernel<<<dim3(32, 8),  tb, 0, stream>>>(P.m1W,  (__bf16*)P.m1T,  FREQN, DD);
    (void)hipFuncSetAttribute((const void*)cogmem_kernel<1>,
                              hipFuncAttributeMaxDynamicSharedMemorySize, (int)SMEM_BYTES);
    cogmem_kernel<1><<<dim3(BB), tb, SMEM_BYTES, stream>>>(P);
  } else {
    (void)hipFuncSetAttribute((const void*)cogmem_kernel<0>,
                              hipFuncAttributeMaxDynamicSharedMemorySize, (int)SMEM_BYTES);
    cogmem_kernel<0><<<dim3(BB), tb, SMEM_BYTES, stream>>>(P);
  }
}
